// LSTMDecoder_80642305949821
// MI455X (gfx1250) — compile-verified, matrix operand order checked by reading, not run on hardware
//
#include <hip/hip_runtime.h>
#include <hip/hip_bf16.h>
#include <math.h>

typedef __attribute__((ext_vector_type(16))) _Float16 v16h;
typedef __attribute__((ext_vector_type(8)))  _Float16 h8;
typedef __attribute__((ext_vector_type(8)))  float    v8f;

#define BB 64
#define NN 512
#define DD 128
#define EE 16384
#define NEGV (-1e9f)
#define KPAD 132   // padded LDS row stride (floats): 16B-aligned, avoids 32-way bank conflicts

__device__ __forceinline__ float sigmoidf_(float x) {
    return 1.0f / (1.0f + __expf(-x));
}

// ---- wave32 butterfly reductions ----
__device__ __forceinline__ float wred_max(float v) {
#pragma unroll
    for (int m = 16; m; m >>= 1) v = fmaxf(v, __shfl_xor(v, m, 32));
    return v;
}
__device__ __forceinline__ float wred_min(float v) {
#pragma unroll
    for (int m = 16; m; m >>= 1) v = fminf(v, __shfl_xor(v, m, 32));
    return v;
}
__device__ __forceinline__ float wred_sum(float v) {
#pragma unroll
    for (int m = 16; m; m >>= 1) v += __shfl_xor(v, m, 32);
    return v;
}
// max value with lowest index on ties (matches jnp.argmax); associative+commutative
__device__ __forceinline__ void wred_argmax(float& v, int& i) {
#pragma unroll
    for (int m = 16; m; m >>= 1) {
        float v2 = __shfl_xor(v, m, 32);
        int   i2 = __shfl_xor(i, m, 32);
        if (v2 > v || (v2 == v && i2 < i)) { v = v2; i = i2; }
    }
}

// ---------------- kernel: f32 -> f16 convert ----------------
__global__ void k_cvt_f16(const float* __restrict__ src, _Float16* __restrict__ dst, int n) {
    int i = blockIdx.x * blockDim.x + threadIdx.x;
    if (i < n) dst[i] = (_Float16)src[i];
}

// ---------------- kernel: zero buffer ----------------
__global__ void k_zero(float* __restrict__ p, int n) {
    int i = blockIdx.x * blockDim.x + threadIdx.x;
    if (i < n) p[i] = 0.0f;
}

// ---------------- kernel: scatter edges into dense A[src,tgt] ----------------
__global__ void k_scatter(const int* __restrict__ eidx, const float* __restrict__ wts,
                          float* __restrict__ A) {
    int e = blockIdx.x * blockDim.x + threadIdx.x;
    if (e < EE) {
        int s = eidx[e];
        int t = eidx[EE + e];
        atomicAdd(&A[(size_t)s * NN + t], wts[e]);
    }
}

// ---------------- kernel: keys = node_emb @ attn_W.T + attn_b via WMMA ----------------
// One wave computes one 16(M)x16(N) f32 tile; K=128 in 4 x v_wmma_f32_16x16x32_f16.
__global__ __launch_bounds__(256)
void k_keys_wmma(const _Float16* __restrict__ embh, const _Float16* __restrict__ wh,
                 const float* __restrict__ attn_b, float* __restrict__ keys) {
    const int wave = threadIdx.x >> 5;     // 0..7 -> N tile (covers D=128)
    const int lane = threadIdx.x & 31;
    const int r    = lane & 15;
    const int hi   = lane >> 4;
    const int m0   = blockIdx.x * 16;
    const int n0   = wave * 16;

    const _Float16* rowA = embh + (size_t)(m0 + r) * DD;
    const _Float16* rowB = wh   + (size_t)(n0 + r) * DD;  // B[k][n] = attn_W[n][k]

    v8f acc = {};
#pragma unroll
    for (int kb = 0; kb < DD; kb += 32) {
        h8 alo = *(const h8*)(rowA + kb + hi * 8);
        h8 ahi = *(const h8*)(rowA + kb + 16 + hi * 8);
        h8 blo = *(const h8*)(rowB + kb + hi * 8);
        h8 bhi = *(const h8*)(rowB + kb + 16 + hi * 8);
        v16h a, b;
#pragma unroll
        for (int i = 0; i < 8; ++i) {
            a[i] = alo[i]; a[i + 8] = ahi[i];
            b[i] = blo[i]; b[i + 8] = bhi[i];
        }
        acc = __builtin_amdgcn_wmma_f32_16x16x32_f16(
            false, a, false, b, (short)0, acc, false, false);
    }

    const float bias = attn_b[n0 + r];
#pragma unroll
    for (int rr = 0; rr < 8; ++rr) {
        int m = m0 + rr + hi * 8;
        keys[(size_t)m * DD + (n0 + r)] = acc[rr] + bias;
    }
}

// ---------------- kernel: persistent per-batch greedy LSTM decode ----------------
// One workgroup per batch element; 512 threads = 16 wave32s; 512 steps in-kernel.
// keys[b] (256 KB, padded to 264 KB) lives in LDS, filled with async-to-LDS loads.
extern __shared__ float s_keys[];   // NN * KPAD floats (dynamic LDS)

__global__ __launch_bounds__(512)
void k_decode(const float* __restrict__ node_emb, const unsigned char* __restrict__ mask,
              const float* __restrict__ edge_weights,
              const float* __restrict__ W_ih, const float* __restrict__ W_hh,
              const float* __restrict__ b_ih, const float* __restrict__ b_hh,
              const float* __restrict__ keys, const float* __restrict__ A,
              const float* __restrict__ p_enc, const float* __restrict__ p_rev,
              const float* __restrict__ p_scw, const float* __restrict__ p_scb,
              float* __restrict__ out) {
    __shared__ __align__(16) float sh[DD], sc[DD], sinp[DD];
    __shared__ __align__(16) float gates[4 * DD];
    __shared__ __align__(16) float scores[NN], prevw[NN], tofir[NN];
    __shared__ unsigned char vis[NN], msk[NN];
    __shared__ float rpart[16];
    __shared__ int   ipart[16];
    __shared__ float s_cmax, s_cmin, s_mx, s_Z;
    __shared__ int s_first, s_prev, s_uniq, s_real, s_curr;

    const int b = blockIdx.x;
    const int t = threadIdx.x;
    const int lane = t & 31;
    const int wid  = t >> 5;
    const float scale = rsqrtf((float)DD);
    const float enc_attn_wt     = p_enc[0];
    const float revisit_penalty = p_rev[0];
    const float shortcut_weight = p_scw[0];
    const float shortcut_bonus  = p_scb[0];

    float* out_tours = out;                          // [B, N+1] as float
    float* out_logp  = out + (size_t)BB * (NN + 1);  // [B, N+1]

    // ---- async fill of keys[b] into padded LDS (ASYNCcnt-tracked, no VGPR round trip) ----
    {
        const float* kb_ = keys + (size_t)b * NN * DD;
#pragma unroll 4
        for (int it = 0; it < 32; ++it) {
            int idx4 = it * 512 + t;            // float4 index within keys[b]
            int row  = idx4 >> 5;               // n
            int c4   = idx4 & 31;               // which float4 of the row
            unsigned loff = (unsigned)(size_t)&s_keys[row * KPAD + c4 * 4];
            const float* g = kb_ + (size_t)idx4 * 4;
            asm volatile("global_load_async_to_lds_b128 %0, %1, off"
                         :: "v"(loff), "v"(g) : "memory");
        }
    }

    // ---- init carry (overlaps with async fill) ----
    vis[t] = 0;
    msk[t] = mask[(size_t)b * NN + t];
    tofir[t] = 0.0f;
    if (t < DD) {
        float s = 0.0f;
        for (int n = 0; n < NN; ++n)
            s += node_emb[((size_t)b * NN + n) * DD + t];
        sinp[t] = s * (1.0f / (float)NN);
        sh[t] = 0.0f;
        sc[t] = 0.0f;
    }
    __syncthreads();
    {
        float v = wred_sum(msk[t] ? 1.0f : 0.0f);
        if (lane == 0) rpart[wid] = v;
        __syncthreads();
        if (t == 0) {
            float r = 0.0f;
            for (int i = 0; i < 16; ++i) r += rpart[i];
            s_real = (int)r;
            s_first = 0; s_prev = 0; s_uniq = 0;
        }
    }
    asm volatile("s_wait_asynccnt 0x0" ::: "memory");
    __syncthreads();

    for (int step = 0; step < NN; ++step) {
        const int prev = s_prev, first = s_first, uniq = s_uniq, realc = s_real;

        // ---- LSTM gates: gates[t] = b_ih[t]+b_hh[t] + W_ih[t,:]·inp + W_hh[t,:]·h ----
        {
            const float4* wi4 = (const float4*)(W_ih + (size_t)t * DD);
            const float4* wh4 = (const float4*)(W_hh + (size_t)t * DD);
            const float4* si4 = (const float4*)sinp;
            const float4* sh4 = (const float4*)sh;
            float acc = b_ih[t] + b_hh[t];
#pragma unroll 8
            for (int q = 0; q < DD / 4; ++q) {
                float4 a = wi4[q], x = si4[q];
                float4 w = wh4[q], y = sh4[q];
                acc += a.x * x.x + a.y * x.y + a.z * x.z + a.w * x.w;
                acc += w.x * y.x + w.y * y.y + w.z * y.z + w.w * y.w;
            }
            gates[t] = acc;
        }
        __syncthreads();
        // ---- cell update (torch gate order i,f,g,o) ----
        if (t < DD) {
            float ig = sigmoidf_(gates[t]);
            float fg = sigmoidf_(gates[DD + t]);
            float gg = tanhf(gates[2 * DD + t]);
            float og = sigmoidf_(gates[3 * DD + t]);
            float cn = fg * sc[t] + ig * gg;
            sc[t] = cn;
            sh[t] = og * tanhf(cn);
        }
        __syncthreads();

        // ---- scores[n] = scale * h·keys_lds[n,:] (+ enc-attn bonus); prev_w row ----
        float sval, pl;
        {
            const float4* kr  = (const float4*)&s_keys[t * KPAD];
            const float4* hh4 = (const float4*)sh;
            float dot = 0.0f;
#pragma unroll 8
            for (int q = 0; q < DD / 4; ++q) {
                float4 kv = kr[q], hv = hh4[q];
                dot += kv.x * hv.x + kv.y * hv.y + kv.z * hv.z + kv.w * hv.w;
            }
            sval = dot * scale;
            if (step > 0) sval += enc_attn_wt * A[(size_t)prev * NN + t];
            pl = edge_weights[((size_t)b * NN + prev) * NN + t];
            scores[t] = sval;
            prevw[t]  = pl;
            // 'first' is fixed after step 0 -> hoist the strided column gather to step 1
            if (step == 1)
                tofir[t] = edge_weights[((size_t)b * NN + t) * NN + first];
        }
        __syncthreads();

        // ---- cmax / cmin over prev_w (wave butterflies + 16-partial combine) ----
        {
            float v = wred_max(pl);
            if (lane == 0) rpart[wid] = v;
            __syncthreads();
            if (t == 0) {
                float r = rpart[0];
                for (int i = 1; i < 16; ++i) r = fmaxf(r, rpart[i]);
                s_cmax = r;
            }
            __syncthreads();
            v = wred_min(pl);
            if (lane == 0) rpart[wid] = v;
            __syncthreads();
            if (t == 0) {
                float r = rpart[0];
                for (int i = 1; i < 16; ++i) r = fminf(r, rpart[i]);
                s_cmin = r;
            }
            __syncthreads();
        }
        const float cmax = s_cmax, cmin = s_cmin;

        // ---- branch logic (revisit / shortcut), masking ----
        {
            const bool  mk     = msk[t] != 0;
            const float direct = prevw[first];
            const float revv   = (vis[t] && mk) ? revisit_penalty : sval;
            const float detour = pl + tofir[t];
            const bool  bm     = mk && (t != first) && (detour < direct);
            const float bonus  = bm ? shortcut_bonus * (direct - detour) / direct : 0.0f;
            const bool  rngok  = cmax > cmin;
            const float normd  = rngok
                ? shortcut_weight * (1.0f - (pl - cmin) / (cmax - cmin)) : 0.0f;
            const float shortc = sval + bonus + normd;
            float s = (uniq < realc) ? revv : (step > 0 ? shortc : sval);
            sval = mk ? s : NEGV;
            scores[t] = sval;
        }

        // ---- greedy argmax (first-index tie-break, matches jnp.argmax) ----
        {
            float v = sval; int i = t;
            wred_argmax(v, i);
            if (lane == 0) { rpart[wid] = v; ipart[wid] = i; }
            __syncthreads();
            if (t == 0) {
                float rv = rpart[0]; int ri = ipart[0];
                for (int k = 1; k < 16; ++k) {
                    float v2 = rpart[k]; int i2 = ipart[k];
                    if (v2 > rv || (v2 == rv && i2 < ri)) { rv = v2; ri = i2; }
                }
                s_mx = rv; s_curr = ri;
            }
            __syncthreads();
        }
        const float mx = s_mx;
        const int curr = s_curr;

        // ---- softmax normalizer (scores[curr]-mx == 0 -> p = 1/Z) ----
        {
            float v = wred_sum(__expf(sval - mx));
            if (lane == 0) rpart[wid] = v;
            __syncthreads();
            if (t == 0) {
                float r = 0.0f;
                for (int i = 0; i < 16; ++i) r += rpart[i];
                s_Z = r;
            }
            __syncthreads();
        }

        if (t == 0) {
            out_tours[(size_t)b * (NN + 1) + step] = (float)curr;
            out_logp [(size_t)b * (NN + 1) + step] = logf(1.0f / s_Z + 1e-10f);
            if (step == 0) s_first = curr;
            if (!vis[curr]) s_uniq = uniq + 1;
            vis[curr] = 1;
            s_prev = curr;
        }
        if (t < DD)
            sinp[t] = node_emb[((size_t)b * NN + curr) * DD + t];
        __syncthreads();
    }

    // ---- final step N: return to first node with log-prob 0 ----
    if (t == 0) {
        out_tours[(size_t)b * (NN + 1) + NN] = (float)s_first;
        out_logp [(size_t)b * (NN + 1) + NN] = 0.0f;
    }
}

extern "C" void kernel_launch(void* const* d_in, const int* in_sizes, int n_in,
                              void* d_out, int out_size, void* d_ws, size_t ws_size,
                              hipStream_t stream) {
    const float*         node_emb = (const float*)d_in[0];
    const unsigned char* mask     = (const unsigned char*)d_in[1];
    const int*           edge_idx = (const int*)d_in[2];
    const float*         attn_wts = (const float*)d_in[3];
    const float*         edge_w   = (const float*)d_in[4];
    const float*         W_ih     = (const float*)d_in[5];
    const float*         W_hh     = (const float*)d_in[6];
    const float*         b_ih     = (const float*)d_in[7];
    const float*         b_hh     = (const float*)d_in[8];
    const float*         attn_W   = (const float*)d_in[9];
    const float*         attn_b   = (const float*)d_in[10];
    const float*         p_enc    = (const float*)d_in[11];
    const float*         p_rev    = (const float*)d_in[12];
    const float*         p_scw    = (const float*)d_in[13];
    const float*         p_scb    = (const float*)d_in[14];

    // workspace layout
    char* ws = (char*)d_ws;
    float*    A    = (float*)(ws);                  // 512*512*4    = 1 MB
    float*    keys = (float*)(ws + (1u << 20));     // 64*512*128*4 = 16 MB
    _Float16* embh = (_Float16*)(ws + (17u << 20)); // 64*512*128*2 = 8 MB
    _Float16* wh   = (_Float16*)(ws + (25u << 20)); // 128*128*2    = 32 KB

    const int nEmb = BB * NN * DD;      // 4194304
    const int nW   = DD * DD;           // 16384
    const int nA   = NN * NN;           // 262144

    k_cvt_f16<<<(nEmb + 255) / 256, 256, 0, stream>>>(node_emb, embh, nEmb);
    k_cvt_f16<<<(nW + 255) / 256, 256, 0, stream>>>(attn_W, wh, nW);
    k_zero<<<(nA + 255) / 256, 256, 0, stream>>>(A, nA);
    k_scatter<<<(EE + 255) / 256, 256, 0, stream>>>(edge_idx, attn_wts, A);
    k_keys_wmma<<<(BB * NN) / 16, 256, 0, stream>>>(embh, wh, attn_b, keys);

    const size_t ldsKeys = (size_t)NN * KPAD * sizeof(float);  // 264 KB dynamic LDS (<320 KB)
    k_decode<<<BB, NN, ldsKeys, stream>>>(node_emb, mask, edge_w,
                                          W_ih, W_hh, b_ih, b_hh,
                                          keys, A,
                                          p_enc, p_rev, p_scw, p_scb,
                                          (float*)d_out);
}
